// Decoder_37752762532693
// MI455X (gfx1250) — compile-verified, matrix operand order checked by reading, not run on hardware
//
#include <hip/hip_runtime.h>
#include <math.h>

// ---------------------------------------------------------------------------
// Decoder with (degenerate) attention + GRU recurrence, B=64, S=512, H=512.
// Attention collapses to uniform weights -> context = mean_s(enc).
// Input path folded into one big bf16 WMMA GEMM (async-LDS double buffered);
// recurrent path is a persistent WMMA kernel with LDS-resident W_hh slices
// and a grid barrier per step.
// ---------------------------------------------------------------------------

typedef __attribute__((ext_vector_type(16))) __bf16 v16bf;
typedef __attribute__((ext_vector_type(8)))  __bf16 v8bf;
typedef __attribute__((ext_vector_type(8)))  float  v8f;

union V16 { v16bf v; v8bf h[2]; };

#define Bb   64
#define Ss   512
#define Hh   512
#define N3H  1536           // 3*H
#define NWG  32             // persistent workgroups in recurrence

// ---- gfx1250 async global->LDS feature detection --------------------------
#if defined(__has_builtin)
#  if __has_builtin(__builtin_amdgcn_global_load_async_to_lds_b128)
#    define ASYNC_LDS 1
#  endif
#  if __has_builtin(__builtin_amdgcn_s_wait_asynccnt)
#    define HAVE_WAIT_ASYNC 1
#  endif
#endif
#ifndef ASYNC_LDS
#  define ASYNC_LDS 0
#endif
#ifndef HAVE_WAIT_ASYNC
#  define HAVE_WAIT_ASYNC 0
#endif

#if ASYNC_LDS
// Builtin expects: (int4 addrspace(1)* src, int4 addrspace(3)* dst, imm off, imm cpol)
typedef int v4i_t __attribute__((vector_size(16)));
typedef __attribute__((address_space(1))) v4i_t g_v4i;
typedef __attribute__((address_space(3))) v4i_t l_v4i;
#endif

static __device__ __forceinline__ float sigmoidf_fast(float x) {
    return 1.0f / (1.0f + __expf(-x));
}

// ---- init: h0 = concat(eh[0], eh[1]);  ctx = mean_s enc -------------------
__global__ __launch_bounds__(256)
void k_h0_ctx(const float* __restrict__ enc, const float* __restrict__ eh,
              float* __restrict__ ctx, float* __restrict__ htf,
              __bf16* __restrict__ htb) {
    int i = blockIdx.x * 256 + threadIdx.x;
    if (i >= Bb * Hh) return;
    int b = i >> 9, h = i & (Hh - 1);
    const float* p = enc + (size_t)b * Ss * Hh + h;
    float sum = 0.f;
    for (int s = 0; s < Ss; ++s) sum += p[(size_t)s * Hh];
    ctx[i] = sum * (1.0f / (float)Ss);
    float h0 = (h < Hh / 2) ? eh[(size_t)b * (Hh / 2) + h]
                            : eh[(size_t)Bb * (Hh / 2) + (size_t)b * (Hh / 2) + (h - Hh / 2)];
    htf[i] = h0;
    htb[i] = (__bf16)h0;
}

// ---- cproj[b,h] = sum_k ctx[b,k]*W_proj[h, H+k] + b_proj[h] ---------------
__global__ __launch_bounds__(256)
void k_cproj(const float* __restrict__ ctx, const float* __restrict__ Wp,
             const float* __restrict__ bp, float* __restrict__ cproj) {
    int i = blockIdx.x * 256 + threadIdx.x;
    if (i >= Bb * Hh) return;
    int b = i >> 9, h = i & (Hh - 1);
    const float* c = ctx + (size_t)b * Hh;
    const float* w = Wp + (size_t)h * (2 * Hh) + Hh;
    float s = bp[h];
    for (int k = 0; k < Hh; ++k) s += c[k] * w[k];
    cproj[i] = s;
}

// ---- bias_gi[b,j] = sum_k cproj[b,k]*W_ih[j,k] + b_ih[j] ------------------
__global__ __launch_bounds__(256)
void k_biasgi(const float* __restrict__ cproj, const float* __restrict__ Wih,
              const float* __restrict__ bih, float* __restrict__ bias_gi) {
    int i = blockIdx.x * 256 + threadIdx.x;
    if (i >= Bb * N3H) return;
    int b = i / N3H, j = i - b * N3H;
    const float* c = cproj + (size_t)b * Hh;
    const float* w = Wih + (size_t)j * Hh;
    float s = bih[j];
    for (int k = 0; k < Hh; ++k) s += c[k] * w[k];
    bias_gi[i] = s;
}

// ---- Wc[j,k] = sum_m W_ih[j,m] * W_proj[m,k]  (k < H), stored bf16 --------
__global__ __launch_bounds__(256)
void k_wc(const float* __restrict__ Wih, const float* __restrict__ Wp,
          __bf16* __restrict__ Wc) {
    int i = blockIdx.x * 256 + threadIdx.x;
    if (i >= N3H * Hh) return;
    int j = i >> 9, k = i & (Hh - 1);
    const float* wi = Wih + (size_t)j * Hh;
    const float* wp = Wp + k;
    float s = 0.f;
    for (int m = 0; m < Hh; ++m) s += wi[m] * wp[(size_t)m * (2 * Hh)];
    Wc[i] = (__bf16)s;
}

// ---- enc f32 -> bf16 -------------------------------------------------------
__global__ __launch_bounds__(256)
void k_enc_bf16(const float* __restrict__ enc, __bf16* __restrict__ encB) {
    size_t i = (size_t)blockIdx.x * 256 + threadIdx.x;
    if (i >= (size_t)Bb * Ss * Hh) return;
    encB[i] = (__bf16)enc[i];
}

__global__ void k_zero(unsigned* c) { if (threadIdx.x == 0) *c = 0u; }

// ---- big parallel GEMM: GI[r, j] = encB[r,:] . Wc[j,:] + bias_gi[b,j] ------
// M = B*S = 32768, N = 1536, K = 512.  WG: 8 waves, tile 128(M) x 64(N).
// B tile (shared by all 8 waves) staged to LDS via async global->LDS copies,
// double buffered across 64-wide K chunks; WMMAs on chunk c overlap the
// async fill of chunk c+1.
__global__ __launch_bounds__(256)
void k_gi(const __bf16* __restrict__ encB, const __bf16* __restrict__ Wc,
          const float* __restrict__ bias_gi, float* __restrict__ GI) {
    constexpr int KC  = 64;   // K chunk
    constexpr int BST = 72;   // padded LDS row stride (elements)
    __shared__ __bf16 Bsm[2 * 64 * BST];

    int tid = threadIdx.x;
    int wave = tid >> 5;
    int lane = tid & 31;
    int ln = lane & 15, lh = lane >> 4;
    int row0 = blockIdx.x * 128 + wave * 16;
    int col0 = blockIdx.y * 64;

    // Cooperative stage of one 64(N) x 64(K) chunk: thread t copies 32 bytes.
    auto stage = [&](int kc, int buf) {
        int r  = tid >> 2;          // 0..63 : N row within tile
        int ko = (tid & 3) * 16;    // 0,16,32,48 : K element offset
        const __bf16* src = Wc + (size_t)(col0 + r) * Hh + kc + ko;
        __bf16* dst = &Bsm[buf * 64 * BST + r * BST + ko];
#if ASYNC_LDS
        __builtin_amdgcn_global_load_async_to_lds_b128((g_v4i*)src,
                                                       (l_v4i*)dst, 0, 0);
        __builtin_amdgcn_global_load_async_to_lds_b128((g_v4i*)src,
                                                       (l_v4i*)dst, 16, 0);
#else
        *(v8bf*)dst       = *(const v8bf*)src;
        *(v8bf*)(dst + 8) = *(const v8bf*)(src + 8);
#endif
    };
    auto wait_stage = []() {
#if ASYNC_LDS
#  if HAVE_WAIT_ASYNC
        __builtin_amdgcn_s_wait_asynccnt(0);
#  else
        asm volatile("s_wait_asynccnt 0x0" ::: "memory");
#  endif
#endif
        __syncthreads();
    };

    v8f acc[4] = {v8f{}, v8f{}, v8f{}, v8f{}};
    const __bf16* arow = encB + (size_t)(row0 + ln) * Hh;

    stage(0, 0);
    wait_stage();
#pragma unroll 1
    for (int c = 0; c < Hh / KC; ++c) {
        int buf = c & 1;
        if (c + 1 < Hh / KC) stage((c + 1) * KC, buf ^ 1);  // async fill next
        const __bf16* bb = &Bsm[buf * 64 * BST];
#pragma unroll
        for (int ks = 0; ks < KC; ks += 32) {
            int ko = c * KC + ks + 8 * lh;
            __builtin_prefetch(arow + ko + 256, 0, 3);      // near prefetch
            V16 A;
            A.h[0] = *(const v8bf*)(arow + ko);
            A.h[1] = *(const v8bf*)(arow + ko + 16);
#pragma unroll
            for (int nt = 0; nt < 4; ++nt) {
                const __bf16* bp = bb + (nt * 16 + ln) * BST + ks + 8 * lh;
                V16 Bf;
                Bf.h[0] = *(const v8bf*)bp;
                Bf.h[1] = *(const v8bf*)(bp + 16);
                acc[nt] = __builtin_amdgcn_wmma_f32_16x16x32_bf16(
                    false, A.v, false, Bf.v, (short)0, acc[nt], false, false);
            }
        }
        wait_stage();   // staged chunk complete + all waves done reading buf
    }

#pragma unroll
    for (int nt = 0; nt < 4; ++nt) {
#pragma unroll
        for (int v = 0; v < 8; ++v) {
            int row = row0 + v + 8 * lh;        // row = b*S + s
            int b = row >> 9;
            int col = col0 + nt * 16 + ln;
            GI[(size_t)row * N3H + col] = acc[nt][v] + bias_gi[(size_t)b * N3H + col];
        }
    }
}

// ---- persistent recurrence: 32 WGs x 128 thr; WG g owns h in [16g,16g+16) -
// Wave w owns batch rows [16w,16w+16). LDS holds the 48x512 bf16 W_hh slice.
__global__ __launch_bounds__(128)
void k_rec(const float* __restrict__ Whh, const float* __restrict__ bhh,
           const float* __restrict__ GI, __bf16* __restrict__ htb,
           float* __restrict__ htf, unsigned* __restrict__ counter,
           float* __restrict__ out) {
    constexpr int LROW = 520;                    // padded row stride (banks)
    __shared__ __bf16 Wsl[48 * LROW];
    int g = blockIdx.x;
    int tid = threadIdx.x;

    // preload + convert W_hh slice (rows: r,z,n gate columns of this h-slice)
    for (int idx = tid; idx < 48 * Hh; idx += 128) {
        int rrow = idx >> 9;
        int cc = idx & (Hh - 1);
        int gate = rrow >> 4;
        int j = gate * Hh + g * 16 + (rrow & 15);
        Wsl[rrow * LROW + cc] = (__bf16)Whh[(size_t)j * Hh + cc];
    }
    __syncthreads();

    int wave = tid >> 5, lane = tid & 31;
    int ln = lane & 15, lh = lane >> 4;
    int h = g * 16 + ln;
    float bhr = bhh[h], bhz = bhh[Hh + h], bhn = bhh[2 * Hh + h];

    for (int s = 0; s < Ss; ++s) {
        int p = s & 1;
        const __bf16* hb = htb + (size_t)p * Bb * Hh;
        const float*  hf = htf + (size_t)p * Bb * Hh;
        __bf16* hbw = htb + (size_t)(1 - p) * Bb * Hh;
        float*  hfw = htf + (size_t)(1 - p) * Bb * Hh;

        v8f ar = {}, az = {}, an = {};
        const __bf16* arow = hb + (size_t)(wave * 16 + ln) * Hh;
        for (int kk = 0; kk < Hh; kk += 32) {
            int ko = kk + 8 * lh;
            V16 A;
            A.h[0] = *(const v8bf*)(arow + ko);
            A.h[1] = *(const v8bf*)(arow + ko + 16);
            const __bf16* b0 = &Wsl[(0 * 16 + ln) * LROW + ko];
            const __bf16* b1 = &Wsl[(1 * 16 + ln) * LROW + ko];
            const __bf16* b2 = &Wsl[(2 * 16 + ln) * LROW + ko];
            V16 B0, B1, B2;
            B0.h[0] = *(const v8bf*)b0; B0.h[1] = *(const v8bf*)(b0 + 16);
            B1.h[0] = *(const v8bf*)b1; B1.h[1] = *(const v8bf*)(b1 + 16);
            B2.h[0] = *(const v8bf*)b2; B2.h[1] = *(const v8bf*)(b2 + 16);
            ar = __builtin_amdgcn_wmma_f32_16x16x32_bf16(false, A.v, false, B0.v, (short)0, ar, false, false);
            az = __builtin_amdgcn_wmma_f32_16x16x32_bf16(false, A.v, false, B1.v, (short)0, az, false, false);
            an = __builtin_amdgcn_wmma_f32_16x16x32_bf16(false, A.v, false, B2.v, (short)0, an, false, false);
        }

#pragma unroll
        for (int v = 0; v < 8; ++v) {
            int b = wave * 16 + v + 8 * lh;
            size_t gbase = ((size_t)b * Ss + s) * N3H;
            float r = sigmoidf_fast(GI[gbase + h]            + ar[v] + bhr);
            float z = sigmoidf_fast(GI[gbase + Hh + h]       + az[v] + bhz);
            float n = tanhf        (GI[gbase + 2 * Hh + h]   + (an[v] + bhn) * r);
            float hp = hf[(size_t)b * Hh + h];
            float hn = (1.0f - z) * n + z * hp;
            out[((size_t)b * Ss + s) * Hh + h] = hn;
            hfw[(size_t)b * Hh + h] = hn;
            hbw[(size_t)b * Hh + h] = (__bf16)hn;
        }

        // grid barrier: release writes, count arrivals, acquire before reuse
        __threadfence();
        __syncthreads();
        if (tid == 0) {
            __hip_atomic_fetch_add(counter, 1u, __ATOMIC_RELEASE, __HIP_MEMORY_SCOPE_AGENT);
            unsigned target = (unsigned)NWG * (unsigned)(s + 1);
            while (__hip_atomic_load(counter, __ATOMIC_ACQUIRE, __HIP_MEMORY_SCOPE_AGENT) < target) {
                __builtin_amdgcn_s_sleep(1);
            }
        }
        __syncthreads();
    }
}

// ---------------------------------------------------------------------------
extern "C" void kernel_launch(void* const* d_in, const int* in_sizes, int n_in,
                              void* d_out, int out_size, void* d_ws, size_t ws_size,
                              hipStream_t stream) {
    (void)in_sizes; (void)n_in; (void)out_size; (void)ws_size;
    const float* enc  = (const float*)d_in[0];   // [B,S,H]
    const float* eh   = (const float*)d_in[1];   // [2,B,H/2]
    const float* Wp   = (const float*)d_in[2];   // [H,2H]
    const float* bp   = (const float*)d_in[3];   // [H]
    const float* Wih  = (const float*)d_in[4];   // [3H,H]
    const float* Whh  = (const float*)d_in[5];   // [3H,H]
    const float* bih  = (const float*)d_in[6];   // [3H]
    const float* bhh  = (const float*)d_in[7];   // [3H]
    float* out = (float*)d_out;                  // [B,S,H]

    char* ws = (char*)d_ws;
    size_t off = 0;
    auto alloc = [&](size_t bytes) -> char* {
        char* p = ws + off;
        off += (bytes + 255) & ~(size_t)255;
        return p;
    };
    unsigned* counter = (unsigned*)alloc(256);
    float*  ctx     = (float*) alloc((size_t)Bb * Hh * 4);
    float*  cproj   = (float*) alloc((size_t)Bb * Hh * 4);
    float*  bias_gi = (float*) alloc((size_t)Bb * N3H * 4);
    float*  htf     = (float*) alloc((size_t)2 * Bb * Hh * 4);
    __bf16* htb     = (__bf16*)alloc((size_t)2 * Bb * Hh * 2);
    __bf16* Wc      = (__bf16*)alloc((size_t)N3H * Hh * 2);
    __bf16* encB    = (__bf16*)alloc((size_t)Bb * Ss * Hh * 2);
    float*  GI      = (float*) alloc((size_t)Bb * Ss * N3H * 4);

    k_zero<<<1, 64, 0, stream>>>(counter);
    k_h0_ctx<<<(Bb * Hh + 255) / 256, 256, 0, stream>>>(enc, eh, ctx, htf, htb);
    k_cproj<<<(Bb * Hh + 255) / 256, 256, 0, stream>>>(ctx, Wp, bp, cproj);
    k_biasgi<<<(Bb * N3H + 255) / 256, 256, 0, stream>>>(cproj, Wih, bih, bias_gi);
    k_wc<<<(N3H * Hh + 255) / 256, 256, 0, stream>>>(Wih, Wp, Wc);
    k_enc_bf16<<<(int)(((size_t)Bb * Ss * Hh + 255) / 256), 256, 0, stream>>>(enc, encB);
    dim3 gi_grid(Bb * Ss / 128, N3H / 64);
    k_gi<<<gi_grid, 256, 0, stream>>>(encB, Wc, bias_gi, GI);
    k_rec<<<NWG, 128, 0, stream>>>(Whh, bhh, GI, htb, htf, counter, out);
}
